// HyperNetworkNCA_70712341561435
// MI455X (gfx1250) — matrix operand order repaired and checked by reading, not compiled, not run on hardware
//
#include <hip/hip_runtime.h>
#include <hip/hip_fp16.h>

#define NB    128     // batch
#define GC    12      // grid channels
#define EMB   64
#define HID   256
#define WSZ   1296    // GC*GC*9
#define POUT  2616    // 2*WSZ + 2*GC
#define HW    64
#define STEPS 32
#define CPAD  16      // channels padded to 16 for WMMA / vector LDS access

typedef __attribute__((ext_vector_type(16))) _Float16 v16h;
typedef __attribute__((ext_vector_type(8)))  float    v8f;

union V16U { float4 f4[2]; v16h h; };
union P8U  { float4 f4; _Float16 h[8]; };

// ---------------------------------------------------------------------------
// Phase 1: hypernetwork. One block per sample, 256 threads (== HID).
// Writes per-sample NCA params [NB, 2616] f32 into workspace.
// ---------------------------------------------------------------------------
__global__ __launch_bounds__(256) void hyper_kernel(
    const float* __restrict__ emb_table, const float* __restrict__ hw1,
    const float* __restrict__ hb1, const float* __restrict__ ln1_g,
    const float* __restrict__ ln1_b, const float* __restrict__ hw2,
    const float* __restrict__ hb2, const float* __restrict__ ln2_g,
    const float* __restrict__ ln2_b, const float* __restrict__ hw3,
    const float* __restrict__ hb3, const float* __restrict__ wscale,
    const int* __restrict__ labels, float* __restrict__ params)
{
  __shared__ float s_in[HID];
  __shared__ float s_red[HID];
  const int b = blockIdx.x, tid = threadIdx.x;

  auto ln_relu = [&](float acc, const float* g, const float* bt) -> float {
    s_red[tid] = acc; __syncthreads();
    for (int s = 128; s > 0; s >>= 1) {
      if (tid < s) s_red[tid] += s_red[tid + s];
      __syncthreads();
    }
    float mean = s_red[0] * (1.f / 256.f);
    __syncthreads();
    float d = acc - mean;
    s_red[tid] = d * d; __syncthreads();
    for (int s = 128; s > 0; s >>= 1) {
      if (tid < s) s_red[tid] += s_red[tid + s];
      __syncthreads();
    }
    float var = s_red[0] * (1.f / 256.f);
    __syncthreads();
    float v = d * rsqrtf(var + 1e-5f) * g[tid] + bt[tid];
    return fmaxf(v, 0.f);
  };

  const int lab = labels[b];
  if (tid < EMB) s_in[tid] = emb_table[lab * EMB + tid];
  __syncthreads();

  // layer 1: [64] -> [256]
  float acc = hb1[tid];
  for (int k = 0; k < EMB; ++k) acc = fmaf(s_in[k], hw1[k * HID + tid], acc);
  float h = ln_relu(acc, ln1_g, ln1_b);
  s_in[tid] = h;
  __syncthreads();

  // layer 2: [256] -> [256]
  acc = hb2[tid];
  for (int k = 0; k < HID; ++k) acc = fmaf(s_in[k], hw2[k * HID + tid], acc);
  h = ln_relu(acc, ln2_g, ln2_b);
  s_in[tid] = h;
  __syncthreads();

  // layer 3: [256] -> [2616], tanh * weight_scale
  const float ws = wscale[0];
  for (int o = tid; o < POUT; o += 256) {
    float a = hb3[o];
    for (int k = 0; k < HID; ++k) a = fmaf(s_in[k], hw3[k * POUT + o], a);
    params[(size_t)b * POUT + o] = tanhf(a) * ws;
  }
}

// ---------------------------------------------------------------------------
// Phase 2: NCA. One block per sample, 256 threads = 8 waves.
// Entire grid + hidden activations resident in LDS (f16, 256 KB) for all
// 32 steps. Convs via v_wmma_f32_16x16x32_f16 implicit GEMM:
//   K ordering: k = tap*16 + cin  (2 taps per 32-wide k-step, 5 k-steps)
//   B operand per lane = 32 contiguous bytes (16 padded channels of 1 pixel)
// B tiles are double-buffered across output tiles so DS latency hides under
// the previous tile's WMMA chain.
// ---------------------------------------------------------------------------
__global__ __launch_bounds__(256) void nca_kernel(
    const float* __restrict__ params, const float* __restrict__ pattern,
    float* __restrict__ out)
{
  extern __shared__ unsigned char smem[];
  _Float16* s_grid = (_Float16*)smem;                       // [64][64][16] f16
  _Float16* s_hid  = (_Float16*)(smem + HW * HW * CPAD * 2);// [64][64][16] f16

  const int b    = blockIdx.x;
  const int tid  = threadIdx.x;
  const int lane = tid & 31;
  const int wave = tid >> 5;
  const int co   = lane & 15;   // output channel row (A matrix M)
  const int hi   = lane >> 4;   // half-wave selector (B: tap pair element)
  const int lnx  = lane & 15;   // pixel column within tile (B matrix N)

  // --- Build per-sample A matrices (f16) + biases, held in VGPRs ---------
  v16h Aw[2][5];
  float bias[2][8];
  const float* pb = params + (size_t)b * POUT;
#pragma unroll
  for (int c = 0; c < 2; ++c) {
    const float* w  = pb + c * (WSZ + GC);
    const float* bb = w + WSZ;
#pragma unroll
    for (int s = 0; s < 5; ++s) {
      v16h a;
#pragma unroll
      for (int e = 0; e < 16; ++e) {
        // A layout: lane<16 -> K {0..15} (tap 2s); lane>=16 -> K {16..31} (tap 2s+1)
        int tap = 2 * s + hi;
        int cin = e;
        float v = 0.f;
        if (tap < 9 && cin < GC && co < GC) v = w[co * 108 + cin * 9 + tap];
        a[e] = (_Float16)v;
      }
      Aw[c][s] = a;
    }
#pragma unroll
    for (int i = 0; i < 8; ++i) {
      int ch = i + 8 * hi;
      bias[c][i] = (ch < GC) ? bb[ch] : 0.f;
    }
  }

  // --- Init grid: zeros + centered 7x7 pattern ----------------------------
  for (int i = tid; i < HW * HW * CPAD; i += 256) s_grid[i] = (_Float16)0.f;
  __syncthreads();
  for (int i = tid; i < GC * 49; i += 256) {
    int c = i / 49, r = i % 49;
    int py = r / 7, px = r % 7;
    s_grid[((28 + py) * HW + (28 + px)) * CPAD + c] = (_Float16)pattern[i];
  }
  __syncthreads();

  // --- Load the 5 B tiles for one 16x16 output tile (10x ds_load_b128) ----
  auto loadB = [&](const _Float16* __restrict__ src, int t, V16U (&bu)[5]) {
    int y  = t >> 2;
    int px = ((t & 3) << 4) + lnx;
    int ry[3], cx[3];
#pragma unroll
    for (int k = 0; k < 3; ++k) {
      ry[k] = ((y + k - 1) & 63) * (HW * CPAD);   // wrapped row offsets
      cx[k] = ((px + k - 1) & 63) * CPAD;         // wrapped col offsets
    }
#pragma unroll
    for (int s = 0; s < 5; ++s) {
      const int tapL = 2 * s;
      const int tapH = (2 * s + 1 > 8) ? 8 : (2 * s + 1);  // kstep4 upper: A=0
      int offL = ry[tapL / 3] + cx[tapL % 3];
      int offH = ry[tapH / 3] + cx[tapH % 3];
      int off  = hi ? offH : offL;                 // one cndmask per k-step
      const float4* p = (const float4*)(src + off);
      bu[s].f4[0] = p[0];
      bu[s].f4[1] = p[1];
    }
  };

  // --- 5 accumulator-chained WMMAs ----------------------------------------
  auto mma5 = [&](const v16h (&A5)[5], const V16U (&bu)[5]) -> v8f {
    v8f acc = {};
#pragma unroll
    for (int s = 0; s < 5; ++s)
      acc = __builtin_amdgcn_wmma_f32_16x16x32_f16(
          false, A5[s], false, bu[s].h, (short)0, acc, false, false);
    return acc;
  };

  auto store_hidden = [&](int t, v8f h) {
    int y  = t >> 2;
    int px = ((t & 3) << 4) + lnx;
    P8U o;
#pragma unroll
    for (int j = 0; j < 8; ++j) {
      float v = h[j] + bias[0][j];
      o.h[j] = (_Float16)(v > 0.f ? v : 0.f);
    }
    *(float4*)(s_hid + (y * HW + px) * CPAD + 8 * hi) = o.f4;
  };

  const size_t FRAME = (size_t)NB * HW * HW;  // 524288 floats per traj frame

  auto update_grid = [&](int t, v8f u, int step) {
    int y  = t >> 2;
    int px = ((t & 3) << 4) + lnx;
    _Float16* gptr = s_grid + (y * HW + px) * CPAD + 8 * hi;
    P8U gold; gold.f4 = *(const float4*)gptr;
    P8U gnew;
    float g0 = 0.f;
#pragma unroll
    for (int j = 0; j < 8; ++j) {
      float gv = (float)gold.h[j];
      float gn = gv + 0.1f * tanhf(u[j] + bias[1][j]);
      gnew.h[j] = (_Float16)gn;
      if (j == 0) g0 = gn;
    }
    *(float4*)gptr = gnew.f4;
    if (hi == 0) {  // this half-wave holds channel 0 (RGB)
      float sig = 1.f / (1.f + expf(-g0));
      size_t pix = (size_t)b * (HW * HW) + y * HW + px;
      out[FRAME + (size_t)step * FRAME + pix] = sig;  // traj[step]
      if (step == STEPS - 1) out[pix] = sig;          // final image
    }
  };

  const int tbase = wave * 32;  // this wave's 32 tiles (8 rows x 4 x-tiles)

  for (int step = 0; step < STEPS; ++step) {
    // ---- conv1: grid -> hidden (relu); double-buffered B ----
    {
      V16U bufA[5], bufB[5];
      loadB(s_grid, tbase, bufA);
      for (int i = 0; i < 32; i += 2) {
        loadB(s_grid, tbase + i + 1, bufB);          // prefetch tile i+1
        v8f h0 = mma5(Aw[0], bufA);                  // waits dscnt<=10
        store_hidden(tbase + i, h0);
        loadB(s_grid, tbase + ((i + 2) & 31), bufA); // prefetch tile i+2
        v8f h1 = mma5(Aw[0], bufB);
        store_hidden(tbase + i + 1, h1);
      }
    }
    __syncthreads();

    // ---- conv2: hidden -> update; g += 0.1*tanh(u); emit sigmoid(ch0) ----
    {
      V16U bufA[5], bufB[5];
      loadB(s_hid, tbase, bufA);
      for (int i = 0; i < 32; i += 2) {
        loadB(s_hid, tbase + i + 1, bufB);
        v8f u0 = mma5(Aw[1], bufA);
        update_grid(tbase + i, u0, step);
        loadB(s_hid, tbase + ((i + 2) & 31), bufA);
        v8f u1 = mma5(Aw[1], bufB);
        update_grid(tbase + i + 1, u1, step);
      }
    }
    __syncthreads();
  }
}

// ---------------------------------------------------------------------------
extern "C" void kernel_launch(void* const* d_in, const int* in_sizes, int n_in,
                              void* d_out, int out_size, void* d_ws, size_t ws_size,
                              hipStream_t stream) {
  const float* emb_table = (const float*)d_in[0];
  const float* hw1       = (const float*)d_in[1];
  const float* hb1       = (const float*)d_in[2];
  const float* ln1_g     = (const float*)d_in[3];
  const float* ln1_b     = (const float*)d_in[4];
  const float* hw2       = (const float*)d_in[5];
  const float* hb2       = (const float*)d_in[6];
  const float* ln2_g     = (const float*)d_in[7];
  const float* ln2_b     = (const float*)d_in[8];
  const float* hw3       = (const float*)d_in[9];
  const float* hb3       = (const float*)d_in[10];
  const float* wscale    = (const float*)d_in[11];
  const float* pattern   = (const float*)d_in[12];
  const int*   labels    = (const int*)d_in[13];

  float* params = (float*)d_ws;  // [NB, 2616] f32

  hyper_kernel<<<NB, 256, 0, stream>>>(emb_table, hw1, hb1, ln1_g, ln1_b,
                                       hw2, hb2, ln2_g, ln2_b, hw3, hb3,
                                       wscale, labels, params);

  // grid f16 (128 KB) + hidden f16 (128 KB) fully LDS-resident
  const int smem_bytes = HW * HW * CPAD * 2 * 2;  // 262144
  (void)hipFuncSetAttribute(reinterpret_cast<const void*>(nca_kernel),
                            hipFuncAttributeMaxDynamicSharedMemorySize,
                            smem_bytes);
  nca_kernel<<<NB, 256, smem_bytes, stream>>>(params, pattern, (float*)d_out);
}
